// GPTBlock_21887153340719
// MI455X (gfx1250) — compile-verified, matrix operand order checked by reading, not run on hardware
//
#include <hip/hip_runtime.h>
#include <hip/hip_bf16.h>

// ---------------------------------------------------------------------------
// GPT block forward for MI455X (gfx1250, wave32, WMMA).
// bf16 operands -> v_wmma_f32_16x16x32_bf16, fp32 accumulation, fp32
// residual path. Tile staging uses GLOBAL_LOAD_ASYNC_TO_LDS_B128 (ASYNCcnt)
// when the toolchain exposes the builtin, else a plain b128 copy.
// ---------------------------------------------------------------------------

typedef __bf16 bf16x16 __attribute__((ext_vector_type(16)));
typedef __bf16 bf16x8  __attribute__((ext_vector_type(8)));
typedef float  f32x8   __attribute__((ext_vector_type(8)));
typedef int    i32x4   __attribute__((vector_size(16)));

#define DIM   1024
#define SEQ   2048
#define NB    2
#define ROWS  (NB * SEQ)   // 4096 token rows
#define NH    16
#define DH    64
#define FFD   4096

#if defined(__gfx1250__) && __has_builtin(__builtin_amdgcn_global_load_async_to_lds_b128)
#define USE_ASYNC_LDS 1
#else
#define USE_ASYNC_LDS 0
#endif

// 16-byte global -> LDS tile copy. Async (ASYNCcnt) when available.
// Builtin prototype (from hipcc diagnostic): first/second params are
// pointers to <4 x i32> in the global / LDS address spaces.
__device__ __forceinline__ void copy16_g2l(const __bf16* g, __bf16* l) {
#if USE_ASYNC_LDS
  __builtin_amdgcn_global_load_async_to_lds_b128(
      (__attribute__((address_space(1))) i32x4*)g,
      (__attribute__((address_space(3))) i32x4*)l, 0, 0);
#else
  *(bf16x8*)l = *(const bf16x8*)g;
#endif
}

// Drain this wave's outstanding async tile loads. Must precede s_barrier:
// the workgroup barrier does not wait on ASYNCcnt.
__device__ __forceinline__ void async_join() {
#if USE_ASYNC_LDS
#if __has_builtin(__builtin_amdgcn_s_wait_asynccnt)
  __builtin_amdgcn_s_wait_asynccnt(0);
#else
  asm volatile("s_wait_asynccnt 0" ::: "memory");
#endif
#endif
}

__device__ __forceinline__ f32x8 wmma_bf16(bf16x16 a, bf16x16 b, f32x8 c) {
  // D = A(16x32 bf16) * B(32x16 bf16) + C(16x16 f32)
  return __builtin_amdgcn_wmma_f32_16x16x32_bf16(false, a, false, b,
                                                 (short)0, c, false, false);
}

// Load a 16x32 bf16 A/B fragment for one lane from a row-major tile
// (LDS or global). Per ISA 16-bit A layout: lanes 0-15 hold K=0..7 and
// 16..23 (VGPR0..7), lanes 16-31 hold K=8..15 and 24..31, row M = lane&15.
__device__ __forceinline__ bf16x16 load_frag(const __bf16* rowptr, int half) {
  union { bf16x16 v; bf16x8 h[2]; } f;
  f.h[0] = *(const bf16x8*)(rowptr + half * 8);
  f.h[1] = *(const bf16x8*)(rowptr + 16 + half * 8);
  return f.v;
}

// ---------------------------------------------------------------------------
// fp32 -> bf16 elementwise conversion (weights / one-off casts)
// ---------------------------------------------------------------------------
__global__ void __launch_bounds__(256) cvt_kernel(const float* __restrict__ in,
                                                  __bf16* __restrict__ out, int n) {
  int i = blockIdx.x * 256 + threadIdx.x;
  if (i < n) out[i] = (__bf16)in[i];
}

// ---------------------------------------------------------------------------
// LayerNorm over last dim (1024), one block per token row, bf16 output.
// ---------------------------------------------------------------------------
__global__ void __launch_bounds__(256) ln_kernel(const float* __restrict__ x,
                                                 const float* __restrict__ w,
                                                 const float* __restrict__ bsh,
                                                 __bf16* __restrict__ out) {
  __shared__ float r1[256];
  __shared__ float r2[256];
  const int t = threadIdx.x;
  const size_t row = blockIdx.x;
  const float* xr = x + row * DIM;
  float xv[4];
  float s = 0.f, s2 = 0.f;
#pragma unroll
  for (int j = 0; j < 4; ++j) {
    xv[j] = xr[t + j * 256];
    s += xv[j];
    s2 += xv[j] * xv[j];
  }
  r1[t] = s; r2[t] = s2;
  __syncthreads();
  for (int off = 128; off > 0; off >>= 1) {
    if (t < off) { r1[t] += r1[t + off]; r2[t] += r2[t + off]; }
    __syncthreads();
  }
  const float mu  = r1[0] * (1.0f / DIM);
  const float var = r2[0] * (1.0f / DIM) - mu * mu;
  const float rs  = rsqrtf(var + 1e-5f);
#pragma unroll
  for (int j = 0; j < 4; ++j) {
    int i = t + j * 256;
    out[row * DIM + i] = (__bf16)((xv[j] - mu) * rs * w[i] + bsh[i]);
  }
}

// ---------------------------------------------------------------------------
// Tiled WMMA GEMM: C[M,N] = A[M,K](bf16) * B[K,N](bf16) + bias
// Block tile 128x128, BK=64 (16 WMMAs between barrier pairs). 8 waves in
// 4(M) x 2(N); each wave owns 32x64. A is staged row-major (async b128 when
// available); B is staged transposed in LDS ([n][k]) so both operand
// fragments are two 16B ds reads. Optional ReLU / fp32 residual / out dtype.
// ---------------------------------------------------------------------------
template <bool RELU, bool RES, bool OUTF32>
__global__ void __launch_bounds__(256) gemm_kernel(const __bf16* __restrict__ A,
                                                   const __bf16* __restrict__ Bm,
                                                   const float* __restrict__ bias,
                                                   const float* __restrict__ resid,
                                                   void* __restrict__ out,
                                                   int M, int N, int K) {
  __shared__ __align__(16) __bf16 As[128][64];
  __shared__ __align__(16) __bf16 Bt[128][64];

  const int t = threadIdx.x;
  const int lane = t & 31, wave = t >> 5;
  const int wm = wave & 3, wn = wave >> 2;
  const int m0 = blockIdx.y * 128, n0 = blockIdx.x * 128;
  const int mh = lane & 15, half = lane >> 4;

  f32x8 acc[2][4];
  const f32x8 zero = {};
#pragma unroll
  for (int mi = 0; mi < 2; ++mi)
#pragma unroll
    for (int f = 0; f < 4; ++f) acc[mi][f] = zero;

  for (int k0 = 0; k0 < K; k0 += 64) {
    __syncthreads();
    {  // A tile: 128 rows x 64 k, direct layout; 2 threads per row.
      int row = t >> 1;
      int seg = (t & 1) * 32;
      const __bf16* src = A + (size_t)(m0 + row) * K + k0 + seg;
#pragma unroll
      for (int j = 0; j < 4; ++j) copy16_g2l(src + j * 8, &As[row][seg + j * 8]);
    }
    {  // B tile transposed: Bt[n][k]. Each thread: two k values, 16 n each.
      int ng = t >> 5;
      int kb = (t & 31) * 2;
#pragma unroll
      for (int kj = 0; kj < 2; ++kj) {
        int kk = kb + kj;
        const __bf16* src = Bm + (size_t)(k0 + kk) * N + n0 + ng * 16;
        bf16x8 b0 = *(const bf16x8*)(src);
        bf16x8 b1 = *(const bf16x8*)(src + 8);
#pragma unroll
        for (int j = 0; j < 8; ++j) {
          Bt[ng * 16 + j][kk]     = b0[j];
          Bt[ng * 16 + 8 + j][kk] = b1[j];
        }
      }
    }
    async_join();
    __syncthreads();

#pragma unroll
    for (int ks = 0; ks < 2; ++ks) {
      bf16x16 a0 = load_frag(&As[wm * 32 + mh][ks * 32], half);
      bf16x16 a1 = load_frag(&As[wm * 32 + 16 + mh][ks * 32], half);
#pragma unroll
      for (int f = 0; f < 4; ++f) {
        bf16x16 bf_ = load_frag(&Bt[wn * 64 + f * 16 + mh][ks * 32], half);
        acc[0][f] = wmma_bf16(a0, bf_, acc[0][f]);
        acc[1][f] = wmma_bf16(a1, bf_, acc[1][f]);
      }
    }
  }

  // Epilogue. C layout: lane -> n = lane&15, row = vgpr_r + 8*(lane>>4).
#pragma unroll
  for (int mi = 0; mi < 2; ++mi) {
#pragma unroll
    for (int f = 0; f < 4; ++f) {
      const int n = n0 + wn * 64 + f * 16 + mh;
      const float bv = bias[n];
#pragma unroll
      for (int r = 0; r < 8; ++r) {
        const int row = m0 + wm * 32 + mi * 16 + r + 8 * half;
        float v = acc[mi][f][r] + bv;
        if (RELU) v = fmaxf(v, 0.0f);
        if (RES) v += resid[(size_t)row * N + n];
        if (OUTF32) ((float*)out)[(size_t)row * N + n] = v;
        else        ((__bf16*)out)[(size_t)row * N + n] = (__bf16)v;
      }
    }
  }
}

// ---------------------------------------------------------------------------
// Attention, faithful to reference: scores masked to literal 0.0 (not -inf)
// BEFORE softmax, so masked keys contribute exp(0 - max) and masked V rows
// participate in the output. We therefore stream over ALL key tiles with an
// online softmax. One block = (batch, head) x 128 queries; each wave owns a
// 16-query tile. Per 32-key tile: 4 WMMA for Q*K^T, 4 WMMA for P*V.
// ---------------------------------------------------------------------------
__global__ void __launch_bounds__(256) attn_kernel(const __bf16* __restrict__ q,
                                                   const __bf16* __restrict__ k,
                                                   const __bf16* __restrict__ v,
                                                   __bf16* __restrict__ outp) {
  __shared__ __align__(16) __bf16 Kt[32][64];     // K tile, [key][d]
  __shared__ __align__(16) __bf16 Vt[64][32];     // V tile transposed [d][key]
  __shared__ float  Sb[8][16][32];                // per-wave raw scores
  __shared__ __align__(16) __bf16 Pb[8][16][32];  // per-wave exp probs (bf16)
  __shared__ float Mrow[8][16], Lrow[8][16], Arow[8][16];

  const int t = threadIdx.x, lane = t & 31, w = t >> 5;
  const int b = blockIdx.x >> 4, h = blockIdx.x & 15;
  const int q0 = blockIdx.y * 128 + w * 16;       // this wave's query base
  const int mh = lane & 15, half = lane >> 4;
  const size_t hd = (size_t)h * DH;

  // Q fragments stay in registers for the whole kernel (16 rows x 64 d).
  const __bf16* qrow = q + ((size_t)(b * SEQ + q0 + mh)) * DIM + hd;
  const bf16x16 aq0 = load_frag(qrow, half);
  const bf16x16 aq1 = load_frag(qrow + 32, half);

  f32x8 acc[4];
  const f32x8 zero = {};
#pragma unroll
  for (int f = 0; f < 4; ++f) acc[f] = zero;

  if (half == 0) { Mrow[w][mh] = -3.0e38f; Lrow[w][mh] = 0.0f; }

  for (int kt = 0; kt < SEQ / 32; ++kt) {
    const int kb0 = kt * 32;
    __syncthreads();
    {  // K tile: [32 keys][64 d], row-major (acts as B^T for Q*K^T).
      int row = t >> 3, seg = (t & 7) * 8;
      const __bf16* src = k + ((size_t)(b * SEQ + kb0 + row)) * DIM + hd + seg;
      copy16_g2l(src, &Kt[row][seg]);
    }
    {  // V tile transposed: Vt[d][key] (acts as B^T for P*V).
      int key = t & 31, dg = (t >> 5) * 8;
      const __bf16* src = v + ((size_t)(b * SEQ + kb0 + key)) * DIM + hd + dg;
      bf16x8 vv = *(const bf16x8*)src;
#pragma unroll
      for (int j = 0; j < 8; ++j) Vt[dg + j][key] = vv[j];
    }
    async_join();
    __syncthreads();

    // --- scores: S(16q x 32k) = Q(16x64) * K^T, two n-halves, two k-steps.
    f32x8 s[2];
#pragma unroll
    for (int f = 0; f < 2; ++f) {
      bf16x16 b0 = load_frag(&Kt[f * 16 + mh][0], half);
      bf16x16 b1 = load_frag(&Kt[f * 16 + mh][32], half);
      f32x8 sz = zero;
      sz = wmma_bf16(aq0, b0, sz);
      s[f] = wmma_bf16(aq1, b1, sz);
    }

    // --- causal mask (literal 0.0, reference semantics) + spill to LDS.
#pragma unroll
    for (int f = 0; f < 2; ++f)
#pragma unroll
      for (int r = 0; r < 8; ++r) {
        const int qg = q0 + r + 8 * half;
        const int kg = kb0 + f * 16 + mh;
        Sb[w][r + 8 * half][f * 16 + mh] = (kg > qg) ? 0.0f : s[f][r];
      }
    asm volatile("s_wait_dscnt 0" ::: "memory");

    // --- online softmax bookkeeping: lane (mh,half) owns half of row mh.
    const float mold = Mrow[w][mh];
    float tmax = -3.0e38f;
    float sv[16];
#pragma unroll
    for (int j = 0; j < 16; ++j) {
      sv[j] = Sb[w][mh][half * 16 + j];
      tmax = fmaxf(tmax, sv[j]);
    }
    tmax = fmaxf(tmax, __shfl_xor(tmax, 16));
    const float mnew = fmaxf(mold, tmax);
    const float al = __expf(mold - mnew);
    float ts = 0.0f;
#pragma unroll
    for (int j = 0; j < 16; ++j) {
      float p = __expf(sv[j] - mnew);
      ts += p;
      Pb[w][mh][half * 16 + j] = (__bf16)p;
    }
    ts += __shfl_xor(ts, 16);
    if (half == 0) {
      Mrow[w][mh] = mnew;
      Arow[w][mh] = al;
      Lrow[w][mh] = Lrow[w][mh] * al + ts;
    }
    asm volatile("s_wait_dscnt 0" ::: "memory");

    // --- rescale running O by alpha(row), then O += P * V.
#pragma unroll
    for (int r = 0; r < 8; ++r) {
      const float a_r = Arow[w][r + 8 * half];
#pragma unroll
      for (int f = 0; f < 4; ++f) acc[f][r] *= a_r;
    }
    const bf16x16 ap = load_frag(&Pb[w][mh][0], half);
#pragma unroll
    for (int f = 0; f < 4; ++f) {
      bf16x16 bv_ = load_frag(&Vt[f * 16 + mh][0], half);
      acc[f] = wmma_bf16(ap, bv_, acc[f]);
    }
  }

  // --- normalize by running sum, write bf16 output at (b, q, h*64 + d).
#pragma unroll
  for (int r = 0; r < 8; ++r) {
    const float li = 1.0f / Lrow[w][r + 8 * half];
    const int qg = q0 + r + 8 * half;
    __bf16* orow = outp + ((size_t)(b * SEQ + qg)) * DIM + hd;
#pragma unroll
    for (int f = 0; f < 4; ++f) orow[f * 16 + mh] = (__bf16)(acc[f][r] * li);
  }
}

// ---------------------------------------------------------------------------
// Host-side orchestration
// ---------------------------------------------------------------------------
extern "C" void kernel_launch(void* const* d_in, const int* in_sizes, int n_in,
                              void* d_out, int out_size, void* d_ws, size_t ws_size,
                              hipStream_t stream) {
  (void)in_sizes; (void)n_in; (void)out_size; (void)ws_size;

  const float* x    = (const float*)d_in[0];
  const float* ln1w = (const float*)d_in[1];
  const float* ln1b = (const float*)d_in[2];
  const float* Wq   = (const float*)d_in[3];
  const float* bq   = (const float*)d_in[4];
  const float* Wk   = (const float*)d_in[5];
  const float* bk   = (const float*)d_in[6];
  const float* Wv   = (const float*)d_in[7];
  const float* bv   = (const float*)d_in[8];
  const float* Wp   = (const float*)d_in[9];
  const float* bp   = (const float*)d_in[10];
  const float* ln2w = (const float*)d_in[11];
  const float* ln2b = (const float*)d_in[12];
  const float* W1   = (const float*)d_in[13];
  const float* b1   = (const float*)d_in[14];
  const float* W2   = (const float*)d_in[15];
  const float* b2   = (const float*)d_in[16];

  char* ws = (char*)d_ws;
  const size_t MB = 1ull << 20;
  __bf16* hb    = (__bf16*)(ws + 0 * MB);    // ln1 out, bf16      8MB
  __bf16* qb    = (__bf16*)(ws + 8 * MB);    // Q                  8MB
  __bf16* kb    = (__bf16*)(ws + 16 * MB);   // K                  8MB
  __bf16* vb    = (__bf16*)(ws + 24 * MB);   // V                  8MB
  __bf16* attnb = (__bf16*)(ws + 32 * MB);   // attention out      8MB
  float*  x1    = (float*)(ws + 40 * MB);    // x + attn proj     16MB
  __bf16* h2b   = (__bf16*)(ws + 56 * MB);   // ln2 out            8MB
  __bf16* ffb   = (__bf16*)(ws + 64 * MB);   // relu(ff1)         32MB
  __bf16* Wqb   = (__bf16*)(ws + 96 * MB);
  __bf16* Wkb   = (__bf16*)(ws + 98 * MB);
  __bf16* Wvb   = (__bf16*)(ws + 100 * MB);
  __bf16* Wpb   = (__bf16*)(ws + 102 * MB);
  __bf16* W1b   = (__bf16*)(ws + 104 * MB);  // 8MB
  __bf16* W2b   = (__bf16*)(ws + 112 * MB);  // 8MB

  auto cvt = [&](const float* src, __bf16* dst, int n) {
    cvt_kernel<<<(n + 255) / 256, 256, 0, stream>>>(src, dst, n);
  };
  cvt(Wq, Wqb, DIM * DIM);
  cvt(Wk, Wkb, DIM * DIM);
  cvt(Wv, Wvb, DIM * DIM);
  cvt(Wp, Wpb, DIM * DIM);
  cvt(W1, W1b, DIM * FFD);
  cvt(W2, W2b, FFD * DIM);

  // h = LN1(x)
  ln_kernel<<<ROWS, 256, 0, stream>>>(x, ln1w, ln1b, hb);

  // Q/K/V projections
  const dim3 gQKV(DIM / 128, ROWS / 128);  // (8, 32)
  gemm_kernel<false, false, false><<<gQKV, 256, 0, stream>>>(hb, Wqb, bq, nullptr, qb, ROWS, DIM, DIM);
  gemm_kernel<false, false, false><<<gQKV, 256, 0, stream>>>(hb, Wkb, bk, nullptr, kb, ROWS, DIM, DIM);
  gemm_kernel<false, false, false><<<gQKV, 256, 0, stream>>>(hb, Wvb, bv, nullptr, vb, ROWS, DIM, DIM);

  // attention (per batch*head, 128 queries per block)
  attn_kernel<<<dim3(NB * NH, SEQ / 128), 256, 0, stream>>>(qb, kb, vb, attnb);

  // x1 = x + attn @ Wp + bp   (fp32)
  gemm_kernel<false, true, true><<<gQKV, 256, 0, stream>>>(attnb, Wpb, bp, x, x1, ROWS, DIM, DIM);

  // h2 = LN2(x1)
  ln_kernel<<<ROWS, 256, 0, stream>>>(x1, ln2w, ln2b, h2b);

  // ff = relu(h2 @ W1 + b1)
  gemm_kernel<true, false, false><<<dim3(FFD / 128, ROWS / 128), 256, 0, stream>>>(
      h2b, W1b, b1, nullptr, ffb, ROWS, FFD, DIM);

  // out = x1 + ff @ W2 + b2   (fp32 -> d_out)
  gemm_kernel<false, true, true><<<gQKV, 256, 0, stream>>>(ffb, W2b, b2, x1, d_out, ROWS, DIM, FFD);
}
